// MutuallyRecursiveRNN_30099130811164
// MI455X (gfx1250) — matrix operand order repaired and checked by reading, not compiled, not run on hardware
//
#include <hip/hip_runtime.h>
#include <hip/hip_bf16.h>

// ---------------------------------------------------------------------------
// MutuallyRecursiveRNN fused kernel for gfx1250 (MI455X, wave32, WMMA).
// 4 waves per workgroup share one 16-batch-row tile and split the 64 output
// columns (wave w owns N-tile jt = w). Weight B-fragments (12 per wave) are
// register-resident for the whole T=512 loop. Per step per wave: 12
// v_wmma_f32_16x16x32_bf16 in three 2-deep accumulate chains (short critical
// path from the freshest state), tanh, and an LDS exchange of the 16x64 state
// (bf16) with real workgroup split-barriers so every wave gets full-width
// A-fragments for the next matmul.
// ---------------------------------------------------------------------------

typedef __attribute__((ext_vector_type(16))) __bf16 bf16x16;
typedef __attribute__((ext_vector_type(8)))  float  f32x8;

#define T_STEPS        512
#define FDIM           64
#define WAVES          4
#define ROWS_PER_BLOCK 16
#define ACT_STRIDE_B   160                  // 64 bf16 (128B) + 32B pad, 16B aligned
#define TILE_BYTES     (16 * ACT_STRIDE_B)  // 2560
#define SMEM_BYTES     (2 * TILE_BYTES)     // tileM + tileP (double buffer)

static __device__ inline f32x8 wmma_bf16(bf16x16 a, bf16x16 b, f32x8 c) {
  return __builtin_amdgcn_wmma_f32_16x16x32_bf16(false, a, false, b,
                                                 (short)0, c, false, false);
}

// Branch-free tanh: prefer native v_tanh_f32 (CDNA5 TRANS op), else exp2+rcp.
static __device__ inline float fast_tanhf(float x) {
#if __has_builtin(__builtin_amdgcn_tanhf)
  return __builtin_amdgcn_tanhf(x);
#elif __has_builtin(__builtin_amdgcn_tanh_f32)
  return __builtin_amdgcn_tanh_f32(x);
#else
  const float xc = fminf(fmaxf(x, -9.0f), 9.0f);
#if __has_builtin(__builtin_amdgcn_exp2f)
  const float e = __builtin_amdgcn_exp2f(xc * 2.88539008177793f);
#else
  const float e = exp2f(xc * 2.88539008177793f);
#endif
#if __has_builtin(__builtin_amdgcn_rcpf)
  return 1.0f - 2.0f * __builtin_amdgcn_rcpf(e + 1.0f);
#else
  return 1.0f - 2.0f / (e + 1.0f);
#endif
#endif
}

static __device__ inline bf16x16 cvt16(const float v[16]) {
  bf16x16 r;
#pragma unroll
  for (int i = 0; i < 16; ++i) r[i] = (__bf16)v[i];
  return r;
}

// WMMA B-fragment (K x N = 32x16) of weight matrix W (row-major [j][k], row
// stride `stride`, column window at colOff) for output tile jt, K-frag kf.
// Lane: n = N index, h = K half. Element e <-> k = kf*32 + 8h + e (e<8), kf*32 + 16 + 8h + (e-8) (e>=8).
static __device__ inline bf16x16 gwfrag(const float* __restrict__ W, int stride,
                                        int colOff, int jt, int kf, int n, int h) {
  const float* s = W + (jt * 16 + n) * stride + colOff + kf * 32 + 8 * h;
  float v[16];
  *(float4*)(v + 0)  = *(const float4*)(s + 0);
  *(float4*)(v + 4)  = *(const float4*)(s + 4);
  *(float4*)(v + 8)  = *(const float4*)(s + 16);
  *(float4*)(v + 12) = *(const float4*)(s + 20);
  return cvt16(v);
}

// WMMA A-fragment straight from the global x row (f32 -> bf16). Lane row M=n;
// element e: two contiguous 8-float runs at k = f*32+8h and k = f*32+16+8h.
static __device__ inline bf16x16 make_xfrag(const float* __restrict__ xr,
                                            int f, int h) {
  const float* c0 = xr + f * 32 + 8 * h;
  const float* c1 = c0 + 16;
  float v[16];
  *(float4*)(v + 0)  = *(const float4*)(c0);
  *(float4*)(v + 4)  = *(const float4*)(c0 + 4);
  *(float4*)(v + 8)  = *(const float4*)(c1);
  *(float4*)(v + 12) = *(const float4*)(c1 + 4);
  return cvt16(v);
}

// A-fragment (bf16) from an LDS tile via two aligned ds_load_b128.
static __device__ inline bf16x16 lds_frag(const unsigned char* c0,
                                          const unsigned char* c1) {
  union U { uint4 u; __bf16 h[8]; } lo, hi;
  lo.u = *(const uint4*)c0;
  hi.u = *(const uint4*)c1;
  bf16x16 r;
#pragma unroll
  for (int i = 0; i < 8; ++i) { r[i] = lo.h[i]; r[i + 8] = hi.h[i]; }
  return r;
}

static __device__ inline bf16x16 afrag(const unsigned char* tile, int f,
                                       int n, int h) {
  const unsigned char* c0 = tile + n * ACT_STRIDE_B + f * 64 + 16 * h;
  return lds_frag(c0, c0 + 32);
}

__global__ __launch_bounds__(WAVES * 32)
void rnn_fused(const float* __restrict__ x,
               const float* __restrict__ mWih, const float* __restrict__ mWhh,
               const float* __restrict__ mbih, const float* __restrict__ mbhh,
               const float* __restrict__ pWih, const float* __restrict__ pWhh,
               const float* __restrict__ pbih, const float* __restrict__ pbhh,
               const float* __restrict__ plW,  const float* __restrict__ plB,
               const float* __restrict__ mlW,  const float* __restrict__ mlB,
               float* __restrict__ out, int B, int T) {
  __shared__ __align__(16) unsigned char smem[SMEM_BYTES];
  unsigned char* tileM = smem;               // m_new exchange buffer
  unsigned char* tileP = smem + TILE_BYTES;  // p_new exchange buffer

  const int tid  = threadIdx.x;
  const int wave = tid >> 5;    // owns N-tile jt = wave
  const int lane = tid & 31;
  const int h    = lane >> 4;   // half-wave (K/M-half selector)
  const int n    = lane & 15;   // N column / M row index
  const int b0   = blockIdx.x * ROWS_PER_BLOCK;
  const int jcol = wave * 16 + n;  // this lane's output column

  // combined biases for this wave's 16 output columns
  const float biasM = mbih[jcol] + mbhh[jcol];
  const float biasP = pbih[jcol] + pbhh[jcol];

  // ---- this wave's 12 weight B-fragments -> registers (loop-invariant) -----
  bf16x16 wmx[2], wmp[2], wmh[2], wpx[2], wpm[2], wph[2];
#pragma unroll
  for (int kf = 0; kf < 2; ++kf) {
    wmx[kf] = gwfrag(mWih, 128, 0,  wave, kf, n, h);
    wmp[kf] = gwfrag(mWih, 128, 64, wave, kf, n, h);
    wmh[kf] = gwfrag(mWhh, 64,  0,  wave, kf, n, h);
    wpx[kf] = gwfrag(pWih, 128, 0,  wave, kf, n, h);
    wpm[kf] = gwfrag(pWih, 128, 64, wave, kf, n, h);
    wph[kf] = gwfrag(pWhh, 64,  0,  wave, kf, n, h);
  }

  const float* xrow = x + (size_t)(b0 + n) * T * FDIM;  // A-layout row M = n

  bf16x16 pf0, pf1, mf0, mf1;  // full-width state A-fragments
#pragma unroll
  for (int i = 0; i < 16; ++i) {
    pf0[i] = (__bf16)0.f; pf1[i] = (__bf16)0.f;
    mf0[i] = (__bf16)0.f; mf1[i] = (__bf16)0.f;
  }

  for (int t = 0; t < T; ++t) {
    const float* xr = xrow + (size_t)t * FDIM;
    const bf16x16 xf0 = make_xfrag(xr, 0, h);
    const bf16x16 xf1 = make_xfrag(xr, 1, h);
    if (wave == 0) __builtin_prefetch(xr + FDIM + 16 * h, 0, 1);

    // ---- metrics cell: m_new = tanh([x,p]·mWih^T + m·mWhh^T + b) ----------
    {
      f32x8 ax, ap, am;
#pragma unroll
      for (int r = 0; r < 8; ++r) { ax[r] = biasM; ap[r] = 0.f; am[r] = 0.f; }
      ax = wmma_bf16(xf0, wmx[0], ax);  // off critical path
      ax = wmma_bf16(xf1, wmx[1], ax);
      am = wmma_bf16(mf0, wmh[0], am);  // m from previous step
      am = wmma_bf16(mf1, wmh[1], am);
      ap = wmma_bf16(pf0, wmp[0], ap);  // freshest state: 2-deep chain only
      ap = wmma_bf16(pf1, wmp[1], ap);
#pragma unroll
      for (int r = 0; r < 8; ++r)
        ((__bf16*)(tileM + (8 * h + r) * ACT_STRIDE_B))[jcol] =
            (__bf16)fast_tanhf((ax[r] + am[r]) + ap[r]);
    }
    __syncthreads();  // waits DScnt then barrier: m_new fully in tileM
    mf0 = afrag(tileM, 0, n, h);
    mf1 = afrag(tileM, 1, n, h);

    // ---- price cell: p_new = tanh([x,m_new]·pWih^T + p·pWhh^T + b) --------
    {
      f32x8 ax, ap, am;
#pragma unroll
      for (int r = 0; r < 8; ++r) { ax[r] = biasP; ap[r] = 0.f; am[r] = 0.f; }
      ax = wmma_bf16(xf0, wpx[0], ax);
      ax = wmma_bf16(xf1, wpx[1], ax);
      ap = wmma_bf16(pf0, wph[0], ap);
      ap = wmma_bf16(pf1, wph[1], ap);
      am = wmma_bf16(mf0, wpm[0], am);  // freshest state: 2-deep chain only
      am = wmma_bf16(mf1, wpm[1], am);
#pragma unroll
      for (int r = 0; r < 8; ++r)
        ((__bf16*)(tileP + (8 * h + r) * ACT_STRIDE_B))[jcol] =
            (__bf16)fast_tanhf((ax[r] + ap[r]) + am[r]);
    }
    __syncthreads();  // p_new fully in tileP
    pf0 = afrag(tileP, 0, n, h);
    pf1 = afrag(tileP, 1, n, h);
  }

  // ---- output heads (post-loop; no cross-wave hazards remain) -------------
  if (wave == 0) {
    // tileP holds the final p (bf16), written before the last barrier
    if (lane < 16) {
      const __bf16* row = (const __bf16*)(tileP + lane * ACT_STRIDE_B);
      float s = plB[0];
#pragma unroll 8
      for (int k = 0; k < 64; ++k) s += (float)row[k] * plW[k];
      out[b0 + lane] = s;  // price_pred [B x 1]
    }
  } else if (wave == 1) {
    // dump this wave's full-width final m A-fragments into tileM (row M = n)
#pragma unroll
    for (int f = 0; f < 2; ++f)
#pragma unroll
      for (int e = 0; e < 16; ++e) {
        const int k = f * 32 + 8 * h + (e < 8 ? e : e + 8);
        ((__bf16*)(tileM + n * ACT_STRIDE_B))[k] = (f == 0) ? mf0[e] : mf1[e];
      }
    // same-wave LDS ops are in-order: safe to read back immediately
    if (lane < 16) {
      const int b = b0 + lane;
      const __bf16* row = (const __bf16*)(tileM + lane * ACT_STRIDE_B);
#pragma unroll
      for (int e = 0; e < 8; ++e) {
        float s = mlB[e];
#pragma unroll 8
        for (int k = 0; k < 64; ++k) s += (float)row[k] * mlW[e * 64 + k];
        out[B + b * 8 + e] = s;  // metrics_pred [B x 8], after price block
      }
    }
  }
}

extern "C" void kernel_launch(void* const* d_in, const int* in_sizes, int n_in,
                              void* d_out, int out_size, void* d_ws, size_t ws_size,
                              hipStream_t stream) {
  const float* x    = (const float*)d_in[0];
  const float* mWih = (const float*)d_in[1];
  const float* mWhh = (const float*)d_in[2];
  const float* mbih = (const float*)d_in[3];
  const float* mbhh = (const float*)d_in[4];
  const float* pWih = (const float*)d_in[5];
  const float* pWhh = (const float*)d_in[6];
  const float* pbih = (const float*)d_in[7];
  const float* pbhh = (const float*)d_in[8];
  const float* plW  = (const float*)d_in[9];
  const float* plB  = (const float*)d_in[10];
  const float* mlW  = (const float*)d_in[11];
  const float* mlB  = (const float*)d_in[12];
  float* out = (float*)d_out;

  const int T = T_STEPS;
  const int B = in_sizes[0] / (T * FDIM);  // 4096

  dim3 grid(B / ROWS_PER_BLOCK);           // 256 blocks x 4 waves = 1024 waves
  dim3 block(WAVES * 32);                  // 128 threads
  rnn_fused<<<grid, block, 0, stream>>>(x, mWih, mWhh, mbih, mbhh,
                                        pWih, pWhh, pbih, pbhh,
                                        plW, plB, mlW, mlB, out, B, T);
}
